// GAT_31267361914934
// MI455X (gfx1250) — matrix operand order repaired and checked by reading, not compiled, not run on hardware
//
#include <hip/hip_runtime.h>
#include <cstdint>
#include <cstddef>

typedef __attribute__((ext_vector_type(16))) __bf16 v16bf;
typedef __attribute__((ext_vector_type(8)))  float  v8f;

union ABf { v16bf v; unsigned u[8]; };
union CF  { v8f   v; float    f[8]; };

constexpr int Bn = 4, Nn = 2048, IN = 256, OUT = 256, H = 4, HD = 64;
constexpr float NEG = 0.2f;

__device__ __forceinline__ unsigned short f2bf(float f) {
  unsigned u = __builtin_bit_cast(unsigned, f);
  u += 0x7FFFu + ((u >> 16) & 1u);            // round-to-nearest-even
  return (unsigned short)(u >> 16);
}
__device__ __forceinline__ float bf2f(unsigned short h) {
  unsigned u = ((unsigned)h) << 16;
  return __builtin_bit_cast(float, u);
}
__device__ __forceinline__ unsigned pack2(unsigned short lo, unsigned short hi) {
  return (unsigned)lo | ((unsigned)hi << 16);
}
__device__ __forceinline__ unsigned packf(float a, float b) { return pack2(f2bf(a), f2bf(b)); }
__device__ __forceinline__ float lrelu(float x) { return fmaxf(x, NEG * x); }

__device__ __forceinline__ void atomicMaxF(float* addr, float val) {
  unsigned* ua = (unsigned*)addr;
  unsigned old = *ua;
  while (__builtin_bit_cast(float, old) < val) {
    unsigned assumed = old;
    old = atomicCAS(ua, assumed, __builtin_bit_cast(unsigned, val));
    if (old == assumed) break;
  }
}

// B-operand (32x16 bf16): lane-half hi holds k=8*hi..8*hi+7 and 16+8*hi..16+8*hi+7; two 16B loads.
__device__ __forceinline__ void loadB(ABf& Bm, const unsigned short* base) {
  *(uint4*)&Bm.u[0] = *(const uint4*)(base);
  *(uint4*)&Bm.u[4] = *(const uint4*)(base + 16);
}

// ---------------------------------------------------------------- weights prep
__global__ __launch_bounds__(256) void k_prep(const float* __restrict__ W,
                                              const float* __restrict__ Wout,
                                              unsigned short* __restrict__ Wt1,
                                              unsigned short* __restrict__ Woutb,
                                              float* __restrict__ sjmax) {
  int c = blockIdx.x, t = threadIdx.x;
  // Wt1[c][i] = W[c/64][i][c%64]  (bf16, B-operand pair-contiguous along i)
  Wt1[c * IN + t]    = f2bf(W[(c >> 6) * (IN * HD) + t * HD + (c & 63)]);
  Woutb[c * OUT + t] = f2bf(Wout[c * OUT + t]);
  if (c == 0 && t < Bn * H) sjmax[t] = -3.0e38f;
}

// ------------------------------------------------------- phase 1: Wh = x @ W
// out: WhT bf16, row (b*256 + h*64 + d), col n  (so attn B-tiles are pair-contiguous)
__global__ __launch_bounds__(256) void k_proj(const float* __restrict__ x,
                                              const unsigned short* __restrict__ Wt1,
                                              unsigned short* __restrict__ WhT) {
  int t = threadIdx.x, w = t >> 5, l = t & 31, hi = l >> 4, m = l & 15;
  int row0 = blockIdx.x * 64 + (w >> 1) * 16;   // flat row in [0,8192)
  int c0   = (w & 1) * 128;
  CF acc[8];
  for (int ct = 0; ct < 8; ct++)
    for (int r = 0; r < 8; r++) acc[ct].f[r] = 0.f;
  const float* xp = x + (size_t)(row0 + m) * IN;
  for (int kk = 0; kk < IN; kk += 32) {
    // A-operand: k = (v<4 ? 2v : 2v+8) + 8*hi  -> four float4 loads
    ABf A;
    float4 x0 = *(const float4*)(xp + kk +      8 * hi);
    float4 x1 = *(const float4*)(xp + kk +  4 + 8 * hi);
    float4 x2 = *(const float4*)(xp + kk + 16 + 8 * hi);
    float4 x3 = *(const float4*)(xp + kk + 20 + 8 * hi);
    A.u[0] = packf(x0.x, x0.y); A.u[1] = packf(x0.z, x0.w);
    A.u[2] = packf(x1.x, x1.y); A.u[3] = packf(x1.z, x1.w);
    A.u[4] = packf(x2.x, x2.y); A.u[5] = packf(x2.z, x2.w);
    A.u[6] = packf(x3.x, x3.y); A.u[7] = packf(x3.z, x3.w);
#pragma unroll
    for (int g = 0; g < 2; g++) {               // 2 groups: preload 4 B, burst 4 WMMA
      ABf Bm[4];
#pragma unroll
      for (int q = 0; q < 4; q++) {
        int c = c0 + (g * 4 + q) * 16 + m;
        loadB(Bm[q], Wt1 + (size_t)c * IN + kk + hi * 8);
      }
#pragma unroll
      for (int q = 0; q < 4; q++)
        acc[g * 4 + q].v = __builtin_amdgcn_wmma_f32_16x16x32_bf16(
            false, A.v, false, Bm[q].v, (short)0, acc[g * 4 + q].v, false, false);
    }
  }
  int b = row0 / Nn;
  int nbase = (row0 % Nn) + hi * 8;
#pragma unroll
  for (int ct = 0; ct < 8; ct++) {
    int c = c0 + ct * 16 + m;
    unsigned short tmp[8];
#pragma unroll
    for (int r = 0; r < 8; r++) tmp[r] = f2bf(acc[ct].f[r]);   // rows n = nbase+r
    *(uint4*)(WhT + (size_t)(b * 256 + c) * Nn + nbase) = *(uint4*)tmp;
  }
}

// ------------------------------------------- phase 2: attention logit stats
__global__ __launch_bounds__(256) void k_sprep(const unsigned short* __restrict__ WhT,
                                               const float* __restrict__ a,
                                               float* __restrict__ s_i,
                                               float* __restrict__ s_j,
                                               float* __restrict__ sjmax) {
  int bh = blockIdx.x >> 3;
  int n  = ((blockIdx.x & 7) << 8) + threadIdx.x;
  float si = 0.f, sj = 0.f;
  for (int d = 0; d < HD; d++) {
    float wh = bf2f(WhT[(size_t)(bh * HD + d) * Nn + n]);
    si += wh * a[d];
    sj += wh * a[HD + d];
  }
  s_i[bh * Nn + n] = si;
  s_j[bh * Nn + n] = sj;
  __shared__ float red[256];
  red[threadIdx.x] = sj;
  __syncthreads();
  for (int s = 128; s > 0; s >>= 1) {
    if (threadIdx.x < s) red[threadIdx.x] = fmaxf(red[threadIdx.x], red[threadIdx.x + s]);
    __syncthreads();
  }
  if (threadIdx.x == 0) atomicMaxF(&sjmax[bh], red[0]);
}

// ---------------- phase 3: masked softmax(leaky(si+sj)) @ Wh   (single sweep)
// softmax shift m = leaky(si + max_j sj) >= every row score (leaky monotone), so no
// per-row max pass is needed; accumulate unnormalized P'@Wh, divide by row-sum at end.
__global__ __launch_bounds__(256) void k_attn(const int* __restrict__ adj,
                                              const unsigned short* __restrict__ WhT,
                                              const float* __restrict__ s_i,
                                              const float* __restrict__ s_j,
                                              const float* __restrict__ sjmax,
                                              unsigned short* __restrict__ multi) {
  __shared__ float sj_s[Nn];
  __shared__ float si_s[128];
  __shared__ __align__(16) unsigned short Bs[2][HD * 32];   // double-buffered Wh chunk
  __shared__ float linv_s[128];
  int t = threadIdx.x, w = t >> 5, l = t & 31, hi = l >> 4, m = l & 15;
  int bh = blockIdx.x >> 4, it = blockIdx.x & 15;
  int b = bh >> 2, h = bh & 3;
  int i0 = it * 128;
  for (int idx = t; idx < Nn; idx += 256) sj_s[idx] = s_j[bh * Nn + idx];
  if (t < 128) si_s[t] = s_i[bh * Nn + i0 + t];
  __syncthreads();
  int iw = i0 + w * 16;                       // this wave's 16 rows
  float si_v = si_s[w * 16 + m];
  float mrow = lrelu(si_v + sjmax[bh]);       // valid softmax shift (upper bound)
  const int* adjp = adj + ((size_t)b * Nn + (iw + m)) * Nn;
  CF acc[4];
  for (int ct = 0; ct < 4; ct++)
    for (int r = 0; r < 8; r++) acc[ct].f[r] = 0.f;
  float lpart = 0.f;

  int buf = 0;
  for (int j0 = 0; j0 < Nn; j0 += 32, buf ^= 1) {
    { // cooperative stage of WhT[bh*64+d][j0:j0+32] into LDS (4 KB)
      int d = t >> 2, seg = t & 3;
      *(uint4*)&Bs[buf][d * 32 + seg * 8] =
          *(const uint4*)(WhT + (size_t)(bh * HD + d) * Nn + j0 + seg * 8);
    }
    __syncthreads();
    if (j0 + 256 < Nn) __builtin_prefetch(adjp + j0 + 256, 0, 0);  // global_prefetch_b8
    // adjacency mask + s_j for this lane's A-layout k positions (k = 2v(+8) + 8*hi)
    int4  a0 = *(const int4*)(adjp + j0 +      8 * hi);
    int4  a1 = *(const int4*)(adjp + j0 +  4 + 8 * hi);
    int4  a2 = *(const int4*)(adjp + j0 + 16 + 8 * hi);
    int4  a3 = *(const int4*)(adjp + j0 + 20 + 8 * hi);
    float4 f0 = *(const float4*)&sj_s[j0 +      8 * hi];
    float4 f1 = *(const float4*)&sj_s[j0 +  4 + 8 * hi];
    float4 f2 = *(const float4*)&sj_s[j0 + 16 + 8 * hi];
    float4 f3 = *(const float4*)&sj_s[j0 + 20 + 8 * hi];
    auto mk = [&](int ax, int ay, float sj0, float sj1) -> unsigned {
      float e0 = ax ? __expf(lrelu(si_v + sj0) - mrow) : 0.f;
      float e1 = ay ? __expf(lrelu(si_v + sj1) - mrow) : 0.f;
      lpart += e0 + e1;
      return packf(e0, e1);
    };
    ABf A;   // unnormalized probability tile (16x32) in A layout
    A.u[0] = mk(a0.x, a0.y, f0.x, f0.y); A.u[1] = mk(a0.z, a0.w, f0.z, f0.w);
    A.u[2] = mk(a1.x, a1.y, f1.x, f1.y); A.u[3] = mk(a1.z, a1.w, f1.z, f1.w);
    A.u[4] = mk(a2.x, a2.y, f2.x, f2.y); A.u[5] = mk(a2.z, a2.w, f2.z, f2.w);
    A.u[6] = mk(a3.x, a3.y, f3.x, f3.y); A.u[7] = mk(a3.z, a3.w, f3.z, f3.w);
    ABf Bm[4];                               // preload all 4 B tiles, burst 4 WMMA
#pragma unroll
    for (int ct = 0; ct < 4; ct++)
      loadB(Bm[ct], &Bs[buf][(ct * 16 + m) * 32 + hi * 8]);
#pragma unroll
    for (int ct = 0; ct < 4; ct++)
      acc[ct].v = __builtin_amdgcn_wmma_f32_16x16x32_bf16(
          false, A.v, false, Bm[ct].v, (short)0, acc[ct].v, false, false);
  }
  // row sums: lane l covers half the K range of row m, partner is l^16
  lpart += __shfl_xor(lpart, 16, 32);
  linv_s[w * 16 + m] = 1.0f / lpart;
  __syncthreads();
#pragma unroll
  for (int ct = 0; ct < 4; ct++) {
#pragma unroll
    for (int r = 0; r < 8; r++) {
      int mr = r + hi * 8;                    // D-layout row
      float o = acc[ct].f[r] * linv_s[w * 16 + mr];
      multi[(size_t)(b * Nn + iw + mr) * OUT + h * HD + ct * 16 + m] = f2bf(o);
    }
  }
}

// --------------------------------------- phase 4: out = multi @ Wout^T + bout
__global__ __launch_bounds__(256) void k_out(const unsigned short* __restrict__ multi,
                                             const unsigned short* __restrict__ Woutb,
                                             const float* __restrict__ bout,
                                             float* __restrict__ out) {
  int t = threadIdx.x, w = t >> 5, l = t & 31, hi = l >> 4, m = l & 15;
  int row0 = blockIdx.x * 64 + (w >> 1) * 16;
  int c0   = (w & 1) * 128;
  CF acc[8];
  for (int ct = 0; ct < 8; ct++)
    for (int r = 0; r < 8; r++) acc[ct].f[r] = 0.f;
  const unsigned short* ap = multi + (size_t)(row0 + m) * OUT;
  for (int kk = 0; kk < OUT; kk += 32) {
    ABf A;   // k = 2v(+8) + 8*hi  -> two 16B loads
    *(uint4*)&A.u[0] = *(const uint4*)(ap + kk +      8 * hi);
    *(uint4*)&A.u[4] = *(const uint4*)(ap + kk + 16 + 8 * hi);
#pragma unroll
    for (int g = 0; g < 2; g++) {
      ABf Bm[4];
#pragma unroll
      for (int q = 0; q < 4; q++) {
        int o = c0 + (g * 4 + q) * 16 + m;
        loadB(Bm[q], Woutb + (size_t)o * OUT + kk + hi * 8);
      }
#pragma unroll
      for (int q = 0; q < 4; q++)
        acc[g * 4 + q].v = __builtin_amdgcn_wmma_f32_16x16x32_bf16(
            false, A.v, false, Bm[q].v, (short)0, acc[g * 4 + q].v, false, false);
    }
  }
#pragma unroll
  for (int ct = 0; ct < 8; ct++) {
    int o = c0 + ct * 16 + m;
    float bb = bout[o];
#pragma unroll
    for (int r = 0; r < 8; r++)
      out[(size_t)(row0 + r + hi * 8) * OUT + o] = acc[ct].f[r] + bb;
  }
}

extern "C" void kernel_launch(void* const* d_in, const int* in_sizes, int n_in,
                              void* d_out, int out_size, void* d_ws, size_t ws_size,
                              hipStream_t stream) {
  (void)in_sizes; (void)n_in; (void)out_size; (void)ws_size;
  const float* x    = (const float*)d_in[0];
  const int*   adj  = (const int*)d_in[1];
  const float* W    = (const float*)d_in[2];
  const float* a    = (const float*)d_in[3];
  const float* Wout = (const float*)d_in[4];
  const float* bout = (const float*)d_in[5];
  char* ws = (char*)d_ws;
  unsigned short* Wt1   = (unsigned short*)(ws + 0);        // 128 KB
  unsigned short* Woutb = (unsigned short*)(ws + 131072);   // 128 KB
  float*          sjmax = (float*)(ws + 262144);            // 64 B (padded 1 KB)
  float*          s_i   = (float*)(ws + 263168);            // 128 KB
  float*          s_j   = (float*)(ws + 394240);            // 128 KB
  unsigned short* WhT   = (unsigned short*)(ws + 525312);   // 4 MB
  unsigned short* multi = (unsigned short*)(ws + 525312 + 4194304); // 4 MB
  float* out = (float*)d_out;

  k_prep <<<256, 256, 0, stream>>>(W, Wout, Wt1, Woutb, sjmax);
  k_proj <<<(Bn * Nn) / 64, 256, 0, stream>>>(x, Wt1, WhT);
  k_sprep<<<Bn * H * (Nn / 256), 256, 0, stream>>>(WhT, a, s_i, s_j, sjmax);
  k_attn <<<Bn * H * (Nn / 128), 256, 0, stream>>>(adj, WhT, s_i, s_j, sjmax, multi);
  k_out  <<<(Bn * Nn) / 64, 256, 0, stream>>>(multi, Woutb, bout, out);
}
